// HypoNerf_28080496181548
// MI455X (gfx1250) — matrix operand order repaired and checked by reading, not compiled, not run on hardware
//
#include <hip/hip_runtime.h>
#include <stdint.h>

// ---------------------------------------------------------------------------
// Fully-fused HypoNerf MLP for gfx1250 (MI455X).
//  Pass 1: prepack per-batch weights to f16, transposed [n][k], zero-padded,
//          plus f32 biases, into d_ws (one-time f32->f16 conversion).
//  Pass 2: fused MLP, 128 points / block, software-pipelined weight staging:
//          GLOBAL_LOAD_ASYNC_TO_LDS_B128 double-buffered in LDS (ASYNCcnt,
//          in-order completion -> s_wait_asynccnt <= outstanding-prefetch),
//          GEMMs with v_wmma_f32_16x16x32_f16, activations ping-pong in LDS.
// ---------------------------------------------------------------------------

typedef __attribute__((ext_vector_type(16))) _Float16 v16h;
typedef __attribute__((ext_vector_type(8)))  _Float16 v8h;
typedef __attribute__((ext_vector_type(8)))  float    v8f;

#define NBATCH   8
#define NPTS     32768
#define HDIM     256
#define TILE_M   128
#define PE_RAW   120                 // 3 * 40 posenc features
#define K0PAD    128                 // layer-0 K padded (120..127 zero)
#define L0_ELEMS (HDIM * K0PAD)      // 32768 halves
#define LH_ELEMS (HDIM * HDIM)       // 65536 halves
#define HD_ELEMS (16 * HDIM)         // 4096 halves (head padded to 16 cols)
#define WPB      (L0_ELEMS + 4 * LH_ELEMS + HD_ELEMS)   // 299008 halves/batch
#define BPB      (5 * HDIM + 4)      // 1284 bias floats / batch

// ---- CDNA5 async global->LDS copy (16B per lane), tracked by ASYNCcnt ----
__device__ __forceinline__ void async_b128(unsigned ldsByteOff, const void* gptr) {
    asm volatile("global_load_async_to_lds_b128 %0, %1, off"
                 :: "v"(ldsByteOff), "v"(gptr) : "memory");
}
template<int N>
__device__ __forceinline__ void wait_async() {          // ASYNCcnt <= N
    asm volatile("s_wait_asynccnt %0" :: "i"(N) : "memory");
}
template<int BYTES>
__device__ __forceinline__ void issue_chunk(unsigned ldsBase, const char* src, int tid) {
    #pragma unroll
    for (int off = 0; off < BYTES; off += 256 * 16)     // BYTES/4096 asyncs/thread
        async_b128(ldsBase + off + tid * 16, src + off + tid * 16);
}

// ---- A fragment (16x32 f16) from row-major LDS act[row][k], ld elems ----
__device__ __forceinline__ v16h load_fragA(const _Float16* p, int rowBase, int kBase,
                                           int ld, int lane) {
    const _Float16* q = p + (rowBase + (lane & 15)) * ld + kBase + ((lane >> 4) << 4);
    v8h lo = *(const v8h*)(q);        // K = kBase + 16*half + 0..7
    v8h hi = *(const v8h*)(q + 8);    // K = kBase + 16*half + 8..15
    return __builtin_shufflevector(lo, hi, 0,1,2,3,4,5,6,7,8,9,10,11,12,13,14,15);
}
// ---- B fragment (32x16 f16) from transposed weights wT[n][k] ----
__device__ __forceinline__ v16h load_fragB(const _Float16* p, int colBase, int kBase,
                                           int ld, int lane) {
    const _Float16* q = p + (colBase + (lane & 15)) * ld + kBase + ((lane >> 4) << 4);
    return *(const v16h*)(q);
}

// ---- compute one 128-output-column chunk: 8 x (16x16) tiles per wave ----
template<int KS>   // K = KS*32 (4 for layer0, 8 for hidden)
__device__ __forceinline__ void compute_chunk(const _Float16* sWbuf, const float* sBias,
                                              _Float16* actOut, const v16h* a,
                                              int c, int lane, int wave)
{
    const int rowBase = wave << 4;
    #pragma unroll
    for (int t = 0; t < 8; ++t) {
        const int nLoc = t << 4;
        v8f acc = {};
        #pragma unroll
        for (int ks = 0; ks < KS; ++ks) {
            v16h bf = load_fragB(sWbuf, nLoc, ks * 32, KS * 32, lane);
            acc = __builtin_amdgcn_wmma_f32_16x16x32_f16(
                    false, a[ks], false, bf, (short)0, acc, false, false);
        }
        const int nIdx = lane & 15;
        const int hl   = lane >> 4;
        const int colG = (c << 7) + nLoc + nIdx;
        const float bs = sBias[colG];
        #pragma unroll
        for (int v = 0; v < 8; ++v) {
            float r = fmaxf(acc[v] + bs, 0.0f);
            actOut[(rowBase + (hl << 3) + v) * HDIM + colG] = (_Float16)r;
        }
    }
}

// ===========================================================================
// Pass 1: prepack weights (f32 -> f16, transpose to [n][k], zero-pad)
// ===========================================================================
__global__ __launch_bounds__(256)
void hyponerf_prepack_kernel(const float* __restrict__ wb0,
                             const float* __restrict__ wb1,
                             const float* __restrict__ wb2,
                             const float* __restrict__ wb3,
                             const float* __restrict__ wb4,
                             const float* __restrict__ rgb,
                             _Float16* __restrict__ wsW,
                             float* __restrict__ wsBias)
{
    const size_t gid = (size_t)blockIdx.x * 256 + threadIdx.x;
    const float* wbh[4] = {wb1, wb2, wb3, wb4};

    if (gid < (size_t)NBATCH * WPB) {
        const int b = (int)(gid / WPB);
        const int r = (int)(gid % WPB);
        float v;
        if (r < L0_ELEMS) {                              // layer 0: wT[n][128]
            const int n = r >> 7, k = r & 127;
            v = (k < PE_RAW) ? wb0[(size_t)b * ((PE_RAW + 1) * HDIM) + (size_t)k * HDIM + n]
                             : 0.0f;
        } else if (r < L0_ELEMS + 4 * LH_ELEMS) {        // hidden: wT[n][256]
            const int rr = r - L0_ELEMS;
            const int l  = rr >> 16;
            const int q  = rr & (LH_ELEMS - 1);
            const int n = q >> 8, k = q & 255;
            v = wbh[l][(size_t)b * ((HDIM + 1) * HDIM) + (size_t)k * HDIM + n];
        } else {                                         // head: wT[16][256]
            const int rr = r - (L0_ELEMS + 4 * LH_ELEMS);
            const int n = rr >> 8, k = rr & 255;
            v = (n < 4) ? rgb[(size_t)b * ((HDIM + 1) * 4) + (size_t)k * 4 + n] : 0.0f;
        }
        wsW[gid] = (_Float16)v;
    } else {
        const size_t t = gid - (size_t)NBATCH * WPB;
        if (t < (size_t)NBATCH * BPB) {
            const int b = (int)(t / BPB);
            const int r = (int)(t % BPB);
            float v;
            if (r < HDIM)
                v = wb0[(size_t)b * ((PE_RAW + 1) * HDIM) + (size_t)PE_RAW * HDIM + r];
            else if (r < 5 * HDIM)
                v = wbh[(r - HDIM) >> 8][(size_t)b * ((HDIM + 1) * HDIM)
                                         + (size_t)HDIM * HDIM + ((r - HDIM) & 255)];
            else
                v = rgb[(size_t)b * ((HDIM + 1) * 4) + (size_t)HDIM * 4 + (r - 5 * HDIM)];
            wsBias[t] = v;
        }
    }
}

// ===========================================================================
// Pass 2: fused MLP with double-buffered async weight pipeline
// ===========================================================================
__global__ __launch_bounds__(256)
void hyponerf_fused_kernel(const float* __restrict__ x,
                           const _Float16* __restrict__ wsW,
                           const float* __restrict__ wsBias,
                           float* __restrict__ out)
{
    __shared__ _Float16 sWb0[128 * HDIM];      // 64 KB weight buffer (even chunks)
    __shared__ _Float16 sWb1[128 * HDIM];      // 64 KB weight buffer (odd chunks)
    __shared__ _Float16 sAct0[TILE_M * HDIM];  // 64 KB
    __shared__ _Float16 sAct1[TILE_M * HDIM];  // 64 KB
    __shared__ float    sBias[HDIM];           // 1 KB        (total 257 KB)

    const int tid  = threadIdx.x;
    const int lane = tid & 31;
    const int wave = tid >> 5;

    const int tilesPerB = NPTS / TILE_M;                 // 256
    const int b     = blockIdx.x / tilesPerB;
    const int pbase = (blockIdx.x % tilesPerB) * TILE_M;

    const _Float16* wB  = wsW + (size_t)b * WPB;
    const float*    bB  = wsBias + (size_t)b * BPB;
    const char* srcL0 = (const char*)wB;                       // 2 x 32 KB chunks
    const char* srcH  = (const char*)(wB + L0_ELEMS);          // 8 x 64 KB chunks
    const char* srcHd = (const char*)(wB + L0_ELEMS + 4 * LH_ELEMS);  // 8 KB

    const unsigned ldsW0 = (unsigned)(uintptr_t)sWb0;
    const unsigned ldsW1 = (unsigned)(uintptr_t)sWb1;

    // prefetch chunk 0 (L0 c0) before doing the posenc transcendentals
    issue_chunk<32768>(ldsW0, srcL0, tid);

    // ---------------- positional encoding -> sAct0[p][0..127] ----------------
    for (int i = tid; i < TILE_M * 60; i += 256) {
        const int p = i / 60;
        const int j = i - p * 60;                        // d*20 + f
        const int d = j / 20;
        const int f = j - d * 20;
        const float xv  = x[((size_t)b * NPTS + pbase + p) * 3 + d];
        const float ang = xv * exp2f((8.0f / 19.0f) * (float)f);
        float sn, cs;
        __sincosf(ang, &sn, &cs);
        sAct0[p * HDIM + j]      = (_Float16)cs;
        sAct0[p * HDIM + 60 + j] = (_Float16)sn;
    }
    for (int i = tid; i < TILE_M * 8; i += 256) {
        const int p = i >> 3;
        sAct0[p * HDIM + PE_RAW + (i & 7)] = (_Float16)0.0f;
    }

    v16h a[8];

    // ---------------- layer 0 (K=128), chunks 0,1 ----------------
    issue_chunk<32768>(ldsW1, srcL0 + 32768, tid);       // prefetch L0 c1
    if (tid < HDIM) sBias[tid] = bB[tid];
    wait_async<8>();                                     // L0 c0 landed
    __syncthreads();                                     // + posenc visible
    #pragma unroll
    for (int ks = 0; ks < 4; ++ks)
        a[ks] = load_fragA(sAct0, wave << 4, ks * 32, HDIM, lane);
    compute_chunk<4>(sWb0, sBias, sAct1, a, 0, lane, wave);
    __syncthreads();

    issue_chunk<65536>(ldsW0, srcH, tid);                // prefetch L1 c0
    wait_async<16>();                                    // L0 c1 landed
    __syncthreads();
    compute_chunk<4>(sWb1, sBias, sAct1, a, 1, lane, wave);
    __syncthreads();

    // ---------------- hidden layers 1..4 (K=256), chunks 2..9 ----------------
    const _Float16* aIn  = sAct1;
    _Float16*       aOut = sAct0;
    for (int l = 0; l < 4; ++l) {
        const char* srcL = srcH + (size_t)l * (LH_ELEMS * 2);
        // even chunk (buf0 holds it; prefetch odd chunk into buf1)
        issue_chunk<65536>(ldsW1, srcL + 65536, tid);
        if (tid < HDIM) sBias[tid] = bB[HDIM + (l << 8) + tid];
        wait_async<16>();
        __syncthreads();
        #pragma unroll
        for (int ks = 0; ks < 8; ++ks)
            a[ks] = load_fragA(aIn, wave << 4, ks * 32, HDIM, lane);
        compute_chunk<8>(sWb0, sBias, aOut, a, 0, lane, wave);
        __syncthreads();
        // odd chunk (buf1; prefetch next layer's even chunk or head into buf0)
        if (l < 3) { issue_chunk<65536>(ldsW0, srcL + 2 * 65536, tid); wait_async<16>(); }
        else       { issue_chunk<8192>(ldsW0, srcHd, tid);             wait_async<2>();  }
        __syncthreads();
        compute_chunk<8>(sWb1, sBias, aOut, a, 1, lane, wave);
        __syncthreads();
        _Float16* t = (_Float16*)aIn; aIn = aOut; aOut = t;
    }
    // aIn == sAct1 holds final hidden activations; head weights landing in buf0

    // ---------------- output head: [128x256] x [256x4] (padded to 16) -------
    if (tid < 4) sBias[tid] = bB[5 * HDIM + tid];
    wait_async<0>();
    __syncthreads();
    {
        const int rowBase = wave << 4;                   // all 8 waves, 16 rows each
        v8f acc = {};
        #pragma unroll
        for (int ks = 0; ks < 8; ++ks) {
            v16h af = load_fragA(aIn, rowBase, ks * 32, HDIM, lane);
            v16h bf = load_fragB(sWb0, 0, ks * 32, HDIM, lane);
            acc = __builtin_amdgcn_wmma_f32_16x16x32_f16(
                    false, af, false, bf, (short)0, acc, false, false);
        }
        const int n  = lane & 15;
        const int hl = lane >> 4;
        if (n < 4) {
            const float bs = sBias[n];
            #pragma unroll
            for (int v = 0; v < 8; ++v) {
                const int row = pbase + rowBase + (hl << 3) + v;
                out[((size_t)b * NPTS + row) * 4 + n] = acc[v] + bs;
            }
        }
    }
}

// ===========================================================================
extern "C" void kernel_launch(void* const* d_in, const int* in_sizes, int n_in,
                              void* d_out, int out_size, void* d_ws, size_t ws_size,
                              hipStream_t stream) {
    (void)in_sizes; (void)n_in; (void)ws_size; (void)out_size;
    const float* x   = (const float*)d_in[0];
    const float* wb0 = (const float*)d_in[1];
    const float* wb1 = (const float*)d_in[2];
    const float* wb2 = (const float*)d_in[3];
    const float* wb3 = (const float*)d_in[4];
    const float* wb4 = (const float*)d_in[5];
    const float* rgb = (const float*)d_in[6];
    float* out = (float*)d_out;

    _Float16* wsW    = (_Float16*)d_ws;                          // 8*WPB halves
    float*    wsBias = (float*)((char*)d_ws + (size_t)NBATCH * WPB * sizeof(_Float16));

    const size_t prepTotal = (size_t)NBATCH * WPB + (size_t)NBATCH * BPB;
    const unsigned prepBlocks = (unsigned)((prepTotal + 255) / 256);
    hyponerf_prepack_kernel<<<dim3(prepBlocks), dim3(256), 0, stream>>>(
        wb0, wb1, wb2, wb3, wb4, rgb, wsW, wsBias);

    dim3 grid(NBATCH * (NPTS / TILE_M));   // 2048 blocks
    dim3 block(256);                       // 8 waves (wave32)
    hyponerf_fused_kernel<<<grid, block, 0, stream>>>(x, wsW, wsBias, out);
}
